// ForwardAttributeGnn_18940805775403
// MI455X (gfx1250) — compile-verified, matrix operand order checked by reading, not run on hardware
//
#include <hip/hip_runtime.h>
#include <hip/hip_bf16.h>

// ---------------------------------------------------------------------------
// ForwardAttributeGnn for MI455X (gfx1250, wave32, WMMA)
// B=8192, A=8, E=512
//   aggregated = concat(src, edge) @ W^T + bias   (split: src@W1^T hoisted)
//   out[:,a,:] = aggregated[:,a,:] @ edge_proj[a]
// Precision: bf16 operands, f32 WMMA accumulation (v_wmma_f32_16x16x32_bf16).
// Async global->LDS staging of the srcAgg tile (ASYNCcnt path), guarded by
// __has_builtin; builtin takes int4 pointers (per hipcc diagnostic).
// ---------------------------------------------------------------------------

typedef __attribute__((ext_vector_type(16))) __bf16 bf16x16;
typedef __attribute__((ext_vector_type(8)))  __bf16 bf16x8;
typedef __attribute__((ext_vector_type(4)))  __bf16 bf16x4;
typedef __attribute__((ext_vector_type(8)))  float  f32x8;
typedef __attribute__((ext_vector_type(4)))  float  f32x4;

#define Bsz 8192
#define Asz 8
#define Esz 512
#define BM  64          // rows per workgroup
#define TN  64          // output cols per wave (8 waves * 64 = 512)

#if defined(__gfx1250__) && \
    __has_builtin(__builtin_amdgcn_global_load_async_to_lds_b128) && \
    __has_builtin(__builtin_amdgcn_s_wait_asynccnt)
#define USE_ASYNC 1
// Builtin parameter types (from hipcc diagnostic): int4 pointers with
// explicit address spaces: global (AS1) source, LDS (AS3) destination.
typedef int v4i_vs __attribute__((vector_size(4 * sizeof(int))));
typedef __attribute__((address_space(1))) v4i_vs* gv4i_p;
typedef __attribute__((address_space(3))) v4i_vs* lv4i_p;
#else
#define USE_ASYNC 0
#endif

__device__ __forceinline__ bf16x16 combine8(bf16x8 lo, bf16x8 hi) {
  return __builtin_shufflevector(lo, hi, 0,1,2,3,4,5,6,7,8,9,10,11,12,13,14,15);
}

// A/B fragment (16x32 / 32x16 bf16) per ISA layout: lanes 0-15 hold
// k {0..7,16..23}; lanes 16-31 hold k {8..15,24..31}; row/col = lane&15.
__device__ __forceinline__ bf16x16 load_frag(const __bf16* p) {
  bf16x8 lo = *(const bf16x8*)(p);
  bf16x8 hi = *(const bf16x8*)(p + 16);
  return combine8(lo, hi);
}

__device__ __forceinline__ f32x8 wmma_bf16(bf16x16 a, bf16x16 b, f32x8 c) {
  return __builtin_amdgcn_wmma_f32_16x16x32_bf16(
      /*neg_a=*/false, a, /*neg_b=*/false, b,
      /*c_mod=*/(short)0, c, /*reuse_a=*/false, /*reuse_b=*/false);
}

// ---------------------------------------------------------------------------
// Kernel 1: convert weights to bf16, N-major (contiguous-K) layouts.
//   W1nk[n][k] = agg_weight[n][k]        (n,k < 512)
//   W2nk[n][k] = agg_weight[n][512+k]
//   Pjk[a][j][i] = edge_proj[a][i][j]    (transposed -> j-major)
// Total elements: 2*262144 + 2097152 = 2621440
// ---------------------------------------------------------------------------
__global__ __launch_bounds__(256) void prep_kernel(
    const float* __restrict__ agg_w, const float* __restrict__ eproj,
    __bf16* __restrict__ W1nk, __bf16* __restrict__ W2nk,
    __bf16* __restrict__ Pjk) {
  int idx = blockIdx.x * 256 + threadIdx.x;
  if (idx < 262144) {
    int n = idx >> 9, k = idx & 511;
    W1nk[idx] = (__bf16)agg_w[n * 1024 + k];
  } else if (idx < 524288) {
    int t = idx - 262144;
    int n = t >> 9, k = t & 511;
    W2nk[t] = (__bf16)agg_w[n * 1024 + 512 + k];
  } else {
    int t = idx - 524288;
    int a = t >> 18;
    int rr = t & 262143;
    int j = rr >> 9, i = rr & 511;
    Pjk[t] = (__bf16)eproj[((size_t)a * Esz + i) * Esz + j];
  }
}

// ---------------------------------------------------------------------------
// Kernel 2: srcAgg[b][n] = bf16( image[b,:] @ W1nk[n,:] + bias[n] )
// GEMM M=8192 N=512 K=512, 64-row tiles, 8 waves x 64 cols.
// ---------------------------------------------------------------------------
__global__ __launch_bounds__(256) void srcagg_kernel(
    const float* __restrict__ image, const __bf16* __restrict__ W1nk,
    const float* __restrict__ bias, __bf16* __restrict__ srcAgg) {
  extern __shared__ __attribute__((aligned(16))) char smem_raw[];
  __bf16* sX = (__bf16*)smem_raw;               // [64][512] bf16 = 64KB

  const int b0  = blockIdx.x * BM;
  const int tid = threadIdx.x;

  // Stage image rows f32 -> bf16 LDS (coalesced float4 loads).
  for (int i4 = tid; i4 < (BM * Esz) / 4; i4 += 256) {
    int e = i4 * 4, row = e >> 9, k = e & 511;
    f32x4 v = *(const f32x4*)(image + (size_t)(b0 + row) * Esz + k);
    bf16x4 o;
    o[0] = (__bf16)v[0]; o[1] = (__bf16)v[1];
    o[2] = (__bf16)v[2]; o[3] = (__bf16)v[3];
    *(bf16x4*)(sX + e) = o;
  }
  __syncthreads();

  const int wave = tid >> 5, lane = tid & 31;
  const int hs = lane >> 4, r = lane & 15;
  const int nb = wave * TN;
  const int kh = hs * 8;

  f32x8 acc[4][4] = {};
  for (int ks = 0; ks < Esz; ks += 32) {
    bf16x16 af[4], bfr[4];
#pragma unroll
    for (int mt = 0; mt < 4; ++mt)
      af[mt] = load_frag(sX + (mt * 16 + r) * Esz + ks + kh);
#pragma unroll
    for (int nt = 0; nt < 4; ++nt)
      bfr[nt] = load_frag(W1nk + (size_t)(nb + nt * 16 + r) * Esz + ks + kh);
#pragma unroll
    for (int nt = 0; nt < 4; ++nt)
#pragma unroll
      for (int mt = 0; mt < 4; ++mt)
        acc[mt][nt] = wmma_bf16(af[mt], bfr[nt], acc[mt][nt]);
  }

  // Epilogue: + bias, store bf16 (C/D layout: VGPR v -> row v + 8*hs, col = r).
#pragma unroll
  for (int mt = 0; mt < 4; ++mt)
#pragma unroll
    for (int nt = 0; nt < 4; ++nt) {
      int c = nb + nt * 16 + r;
      float bv = bias[c];
#pragma unroll
      for (int v = 0; v < 8; ++v) {
        int m = mt * 16 + v + 8 * hs;
        srcAgg[(size_t)(b0 + m) * Esz + c] = (__bf16)(acc[mt][nt][v] + bv);
      }
    }
}

// ---------------------------------------------------------------------------
// Kernel 3 (fused): per (64-row tile, edge slot a):
//   stage edge f32->bf16 LDS (VALU convert) + srcAgg tile via ASYNC->LDS;
//   GEMM1: edge @ W2^T  -> accumulate onto srcAgg tile (LDS, bf16);
//   GEMM2: aggTile @ edge_proj[a] -> f32 global stores.
// LDS: sX 64KB + sAgg 64KB = 128KB (of 320KB/WGP) -> 2 blocks/WGP.
// ---------------------------------------------------------------------------
__global__ __launch_bounds__(256) void fused_kernel(
    const float* __restrict__ edge,      // [B, A, E] f32
    const __bf16* __restrict__ W2nk,     // [E][E] bf16, n-major
    const __bf16* __restrict__ Pjk,      // [A][E][E] bf16, j-major
    const __bf16* __restrict__ srcAgg,   // [B][E] bf16
    float* __restrict__ out) {           // [B, A, E] f32
  extern __shared__ __attribute__((aligned(16))) char smem_raw[];
  __bf16* sX   = (__bf16*)smem_raw;            // [64][512]
  __bf16* sAgg = sX + BM * Esz;                // [64][512]

  const int b0  = blockIdx.x * BM;
  const int a   = blockIdx.y;
  const int tid = threadIdx.x;

  // Phase 0a: async raw copy of the srcAgg bf16 tile straight into LDS
  // (no VGPR round-trip, tracked by ASYNCcnt), overlapping with 0b below.
#if USE_ASYNC
  {
    __bf16* g = const_cast<__bf16*>(srcAgg) + (size_t)b0 * Esz;
    for (int i = tid; i < (BM * Esz) / 8; i += 256) {
      __builtin_amdgcn_global_load_async_to_lds_b128(
          (gv4i_p)(g + (size_t)i * 8),
          (lv4i_p)(sAgg + (size_t)i * 8),
          0, 0);
    }
  }
#else
  {
    const bf16x8* sa = (const bf16x8*)(srcAgg + (size_t)b0 * Esz);
    bf16x8* da = (bf16x8*)sAgg;
    for (int i = tid; i < (BM * Esz) / 8; i += 256) da[i] = sa[i];
  }
#endif

  // Phase 0b: stage edge rows f32 -> bf16 into sX (needs VALU conversion).
  for (int i4 = tid; i4 < (BM * Esz) / 4; i4 += 256) {
    int e = i4 * 4, row = e >> 9, k = e & 511;
    f32x4 v = *(const f32x4*)(edge + ((size_t)(b0 + row) * Asz + a) * Esz + k);
    bf16x4 o;
    o[0] = (__bf16)v[0]; o[1] = (__bf16)v[1];
    o[2] = (__bf16)v[2]; o[3] = (__bf16)v[3];
    *(bf16x4*)(sX + e) = o;
  }

#if USE_ASYNC
  __builtin_amdgcn_s_wait_asynccnt(0);
#endif
  __syncthreads();

  const int wave = tid >> 5, lane = tid & 31;
  const int hs = lane >> 4, r = lane & 15;
  const int nb = wave * TN;
  const int kh = hs * 8;

  // ---- GEMM 1: edge @ W2^T ----
  f32x8 acc[4][4] = {};
  for (int ks = 0; ks < Esz; ks += 32) {
    bf16x16 af[4], bfr[4];
#pragma unroll
    for (int mt = 0; mt < 4; ++mt)
      af[mt] = load_frag(sX + (mt * 16 + r) * Esz + ks + kh);
#pragma unroll
    for (int nt = 0; nt < 4; ++nt)
      bfr[nt] = load_frag(W2nk + (size_t)(nb + nt * 16 + r) * Esz + ks + kh);
#pragma unroll
    for (int nt = 0; nt < 4; ++nt)
#pragma unroll
      for (int mt = 0; mt < 4; ++mt)
        acc[mt][nt] = wmma_bf16(af[mt], bfr[nt], acc[mt][nt]);
  }

  // Epilogue 1: sAgg[m][c] = bf16( f32(sAgg[m][c]) + acc ).
  // Each wave owns a disjoint 64-column stripe -> race-free.
#pragma unroll
  for (int mt = 0; mt < 4; ++mt)
#pragma unroll
    for (int nt = 0; nt < 4; ++nt) {
      int c = nb + nt * 16 + r;
#pragma unroll
      for (int v = 0; v < 8; ++v) {
        int m = mt * 16 + v + 8 * hs;
        int idx = m * Esz + c;
        sAgg[idx] = (__bf16)((float)sAgg[idx] + acc[mt][nt][v]);
      }
    }
  __syncthreads();

  // ---- GEMM 2: aggTile @ edge_proj[a] ----
  const __bf16* Pp = Pjk + (size_t)a * Esz * Esz;
  f32x8 acc2[4][4] = {};
  for (int ks = 0; ks < Esz; ks += 32) {
    bf16x16 af[4], bfr[4];
#pragma unroll
    for (int mt = 0; mt < 4; ++mt)
      af[mt] = load_frag(sAgg + (mt * 16 + r) * Esz + ks + kh);
#pragma unroll
    for (int nt = 0; nt < 4; ++nt)
      bfr[nt] = load_frag(Pp + (size_t)(nb + nt * 16 + r) * Esz + ks + kh);
#pragma unroll
    for (int nt = 0; nt < 4; ++nt)
#pragma unroll
      for (int mt = 0; mt < 4; ++mt)
        acc2[mt][nt] = wmma_bf16(af[mt], bfr[nt], acc2[mt][nt]);
  }

  // Epilogue 2: f32 stores to out[b, a, :].
#pragma unroll
  for (int mt = 0; mt < 4; ++mt)
#pragma unroll
    for (int nt = 0; nt < 4; ++nt) {
      int c = nb + nt * 16 + r;
#pragma unroll
      for (int v = 0; v < 8; ++v) {
        int m = mt * 16 + v + 8 * hs;
        out[((size_t)(b0 + m) * Asz + a) * Esz + c] = acc2[mt][nt][v];
      }
    }
}

// ---------------------------------------------------------------------------
extern "C" void kernel_launch(void* const* d_in, const int* in_sizes, int n_in,
                              void* d_out, int out_size, void* d_ws, size_t ws_size,
                              hipStream_t stream) {
  const float* image = (const float*)d_in[0];   // [B, E]
  const float* edge  = (const float*)d_in[1];   // [B, A, E]
  const float* aggw  = (const float*)d_in[2];   // [E, 2E]
  const float* aggb  = (const float*)d_in[3];   // [E]
  const float* eproj = (const float*)d_in[4];   // [A, E, E]
  float* out = (float*)d_out;

  char* ws = (char*)d_ws;
  const size_t W_BYTES = (size_t)Esz * Esz * 2;           // 512 KB
  __bf16* W1nk   = (__bf16*)(ws);
  __bf16* W2nk   = (__bf16*)(ws + W_BYTES);
  __bf16* Pjk    = (__bf16*)(ws + 2 * W_BYTES);           // 4 MB
  __bf16* srcAgg = (__bf16*)(ws + 2 * W_BYTES + (size_t)Asz * W_BYTES); // 8 MB
  // total workspace use: 13 MB

  prep_kernel<<<10240, 256, 0, stream>>>(aggw, eproj, W1nk, W2nk, Pjk);
  srcagg_kernel<<<Bsz / BM, 256, 64 * 1024, stream>>>(image, W1nk, aggb, srcAgg);
  fused_kernel<<<dim3(Bsz / BM, Asz), 256, 128 * 1024, stream>>>(
      edge, W2nk, Pjk, srcAgg, out);
}
